// BEVFormerBlock_52948356825171
// MI455X (gfx1250) — matrix operand order repaired and checked by reading, not compiled, not run on hardware
//
#include <hip/hip_runtime.h>
#include <hip/hip_bf16.h>
#include <math.h>

// ---------------------------------------------------------------------------
// BEVFormer encoder layer for MI455X (gfx1250, wave32, WMMA).
// Dense projections: LDS-staged GEMM, 128x64 macro-tile per 8-wave workgroup,
// 32x32 per-wave tile -> 4x v_wmma_f32_16x16x32_f16 per K-step.
// A tile staged via GLOBAL_LOAD_ASYNC_TO_LDS_B128 (ASYNCcnt-tracked DMA) when
// the toolchain exposes the builtin, else a synchronous fill with the same
// LDS layout. Deformable sampling: one wave per (q, head), lane = head-dim.
// ---------------------------------------------------------------------------

typedef __attribute__((ext_vector_type(16))) _Float16 v16h;
typedef __attribute__((ext_vector_type(8)))  _Float16 v8h;
typedef __attribute__((ext_vector_type(4)))  _Float16 v4h;
typedef __attribute__((ext_vector_type(8)))  float    v8f;
typedef __attribute__((ext_vector_type(4)))  float    v4f;
typedef __attribute__((ext_vector_type(4)))  int      v4i;

#define NQ_    2500
#define EMBED_ 256
#define HEADS_ 8
#define HD_    32
#define TOT_   5440
#define NCAM_  6

#define BM 128
#define BN 64
#define KS 32
#define LDAF 40  // floats per A row in LDS (16B-aligned chunks, spreads banks)
#define LDB  40  // halfs per Bt row in LDS (B stored transposed: [n][k])

#if defined(__gfx1250__) && __has_builtin(__builtin_amdgcn_global_load_async_to_lds_b128) && __has_builtin(__builtin_amdgcn_s_wait_asynccnt)
#define ASYNC_LDS 1
typedef __attribute__((address_space(1))) v4i* g_v4i_ptr;
typedef __attribute__((address_space(3))) v4i* l_v4i_ptr;
#else
#define ASYNC_LDS 0
#endif

// ------------------------- LDS-staged WMMA GEMM ----------------------------
// C[M,N] = act(A[M,K] @ W[K,N] + bias[N]) (+ resid[M,N])
// A rows < split come from A0, rows >= split from A1 (batch-concat support).
// Requires: K % 32 == 0, N % 64 == 0.
__global__ __launch_bounds__(256)
void gemm_wmma_f16(const float* __restrict__ A0,
                   const float* __restrict__ A1, int split,
                   const float* __restrict__ W,
                   const float* __restrict__ bias,
                   const float* __restrict__ resid,
                   float* __restrict__ C,
                   int M, int N, int K, int relu)
{
  __shared__ float    Af[BM * LDAF];  // 20480 B (A tile, f32)
  __shared__ _Float16 Bt[BN * LDB];   //  5120 B (B tile, f16, transposed)

  const int t    = threadIdx.x;
  const int lane = t & 31;
  const int wave = t >> 5;        // 0..7
  const int wm   = wave >> 1;     // 0..3  (M direction)
  const int wn   = wave & 1;      // 0..1  (N direction)
  const int half = lane >> 4;
  const int l16  = lane & 15;
  const int mbase = blockIdx.y * BM;
  const int nbase = blockIdx.x * BN;

  v8f c[2][2] = {};

  for (int k0 = 0; k0 < K; k0 += KS) {
    // ---- stage A tile (BM x 32 f32), 16B chunks, async DMA when available --
#pragma unroll
    for (int i = 0; i < 4; ++i) {
      int slot = t + i * 256;          // 1024 float4 slots
      int row  = slot >> 3;            // 0..127
      int cq   = (slot & 7) * 4;       // 0,4,...,28
      int gm = mbase + row; if (gm >= M) gm = M - 1;
      const float* Ar = (gm < split) ? (A0 + (size_t)gm * K)
                                     : (A1 + (size_t)(gm - split) * K);
      const float* src = Ar + k0 + cq;
      float* dst = &Af[row * LDAF + cq];
#if ASYNC_LDS
      __builtin_amdgcn_global_load_async_to_lds_b128(
          (g_v4i_ptr)src, (l_v4i_ptr)dst, 0, 0);
#else
      *(v4f*)dst = *(const v4f*)src;
#endif
    }
    // ---- stage B tile transposed (32 x BN f32 -> Bt[n][k] f16) ----
#pragma unroll
    for (int i = 0; i < 2; ++i) {
      int slot = t + i * 256;          // 512 float4 slots
      int kr   = slot >> 4;            // 0..31
      int cq   = (slot & 15) * 4;      // 0,4,...,60
      v4f fv = *(const v4f*)(W + (size_t)(k0 + kr) * N + nbase + cq);
#pragma unroll
      for (int j = 0; j < 4; ++j) Bt[(cq + j) * LDB + kr] = (_Float16)fv[j];
    }
#if ASYNC_LDS
    __builtin_amdgcn_s_wait_asynccnt(0);
#endif
    __syncthreads();

    // ---- fragments from LDS ----
    // A 16x32 f16 layout (ISA 7.12.2): lane half h, chunk g: k = 16g + 8h + e
    // -> two contiguous runs of 8 floats per lane; convert f32->f16 here.
    v16h afr[2], bfr[2];
#pragma unroll
    for (int mt = 0; mt < 2; ++mt) {
      const float* ap = &Af[(wm * 32 + mt * 16 + l16) * LDAF + half * 8];
      v4f a0 = *(const v4f*)(ap);
      v4f a1 = *(const v4f*)(ap + 4);
      v4f a2 = *(const v4f*)(ap + 16);
      v4f a3 = *(const v4f*)(ap + 20);
#pragma unroll
      for (int e = 0; e < 4; ++e) {
        afr[mt][e]      = (_Float16)a0[e];
        afr[mt][4 + e]  = (_Float16)a1[e];
        afr[mt][8 + e]  = (_Float16)a2[e];
        afr[mt][12 + e] = (_Float16)a3[e];
      }
    }
    // B 32x16 f16 layout: lane half h, chunk g: k = 16g + 8h + e; n = l16
#pragma unroll
    for (int nt = 0; nt < 2; ++nt) {
      const _Float16* bp = &Bt[(wn * 32 + nt * 16 + l16) * LDB + half * 8];
      v8h lo = *(const v8h*)(bp);
      v8h hi = *(const v8h*)(bp + 16);
#pragma unroll
      for (int e = 0; e < 8; ++e) { bfr[nt][e] = lo[e]; bfr[nt][8 + e] = hi[e]; }
    }
#pragma unroll
    for (int mt = 0; mt < 2; ++mt)
#pragma unroll
      for (int nt = 0; nt < 2; ++nt)
        c[mt][nt] = __builtin_amdgcn_wmma_f32_16x16x32_f16(
            false, afr[mt], false, bfr[nt], (short)0, c[mt][nt], false, false);
    __syncthreads();
  }

  // ---- epilogue: C/D layout VGPR r -> M = r + 8*half, N = l16 ----
  const bool full = (mbase + BM) <= M;
#pragma unroll
  for (int mt = 0; mt < 2; ++mt) {
#pragma unroll
    for (int nt = 0; nt < 2; ++nt) {
      const int n = nbase + wn * 32 + nt * 16 + l16;
#pragma unroll
      for (int r = 0; r < 8; ++r) {
        const int m = mbase + wm * 32 + mt * 16 + half * 8 + r;
        if (full || m < M) {
          float val = c[mt][nt][r] + bias[n];
          if (relu) val = fmaxf(val, 0.f);
          if (resid) val += resid[(size_t)m * N + n];
          C[(size_t)m * N + n] = val;
        }
      }
    }
  }
}

// ------------------------- elementwise helpers ------------------------------
__global__ void build_qa_kernel(const float* __restrict__ prev,
                                const float* __restrict__ q,
                                const float* __restrict__ pos,
                                float* __restrict__ qa)
{
  int i = blockIdx.x * blockDim.x + threadIdx.x;   // over NQ_*256
  if (i >= NQ_ * EMBED_) return;
  int r = i >> 8, col = i & 255;
  qa[(size_t)r * 512 + col]       = prev[i];
  qa[(size_t)r * 512 + 256 + col] = q[i] + pos[i];
}

__global__ void add_kernel(const float* __restrict__ a,
                           const float* __restrict__ b,
                           float* __restrict__ out, int n)
{
  int i = blockIdx.x * blockDim.x + threadIdx.x;
  if (i < n) out[i] = a[i] + b[i];
}

__global__ void softmax_rows_kernel(float* __restrict__ x, int rows, int cols)
{
  int r = blockIdx.x * blockDim.x + threadIdx.x;
  if (r >= rows) return;
  float* p = x + (size_t)r * cols;
  float mx = -1e30f;
  for (int i = 0; i < cols; ++i) mx = fmaxf(mx, p[i]);
  float s = 0.f;
  for (int i = 0; i < cols; ++i) { float e = __expf(p[i] - mx); p[i] = e; s += e; }
  float inv = 1.f / s;
  for (int i = 0; i < cols; ++i) p[i] *= inv;
}

// one wave32 per row of 256
__global__ void layernorm_kernel(const float* __restrict__ x,
                                 const float* __restrict__ g,
                                 const float* __restrict__ b,
                                 float* __restrict__ y, int rows)
{
  int wid  = threadIdx.x >> 5;
  int lane = threadIdx.x & 31;
  int row  = blockIdx.x * 8 + wid;
  if (row >= rows) return;
  const float* xr = x + (size_t)row * EMBED_;
  float v[8]; float s = 0.f;
#pragma unroll
  for (int i = 0; i < 8; ++i) { v[i] = xr[lane + i * 32]; s += v[i]; }
#pragma unroll
  for (int m = 16; m > 0; m >>= 1) s += __shfl_xor(s, m, 32);
  float mean = s * (1.f / 256.f);
  float vs = 0.f;
#pragma unroll
  for (int i = 0; i < 8; ++i) { float d = v[i] - mean; vs += d * d; }
#pragma unroll
  for (int m = 16; m > 0; m >>= 1) vs += __shfl_xor(vs, m, 32);
  float inv = rsqrtf(vs * (1.f / 256.f) + 1e-5f);
  float* yr = y + (size_t)row * EMBED_;
#pragma unroll
  for (int i = 0; i < 8; ++i) {
    int cidx = lane + i * 32;
    yr[cidx] = (v[i] - mean) * inv * g[cidx] + b[cidx];
  }
}

// ------------------------- deformable sampling ------------------------------
__device__ __forceinline__ float bilin_gather(const float* __restrict__ v,
                                              float x, float y, int Wd, int Hd,
                                              int base, int doff)
{
  float xf = floorf(x), yf = floorf(y);
  int   x0 = (int)xf,   y0 = (int)yf;
  float wx = x - xf,    wy = y - yf;
  float s = 0.f;
#pragma unroll
  for (int cc = 0; cc < 4; ++cc) {
    int xi = x0 + (cc & 1);
    int yi = y0 + (cc >> 1);
    float w = ((cc & 1) ? wx : 1.f - wx) * ((cc >> 1) ? wy : 1.f - wy);
    if (xi >= 0 && xi < Wd && yi >= 0 && yi < Hd)
      s += w * v[(size_t)(base + yi * Wd + xi) * EMBED_ + doff];
  }
  return s;
}

// one wave per (q, head); lane = head-dim
__global__ void tsa_sample_kernel(const float* __restrict__ v,    // (2,NQ,8,32)
                                  const float* __restrict__ off,  // (NQ,128)
                                  const float* __restrict__ aw,   // (NQ,64) softmaxed
                                  const float* __restrict__ ref2d,// (NQ,2)
                                  float* __restrict__ out)        // (NQ,256)
{
  int wid  = (blockIdx.x * blockDim.x + threadIdx.x) >> 5;
  int lane = threadIdx.x & 31;
  if (wid >= NQ_ * HEADS_) return;
  int q = wid >> 3, h = wid & 7;
  float rx = ref2d[q * 2 + 0] * 50.f - 0.5f;
  float ry = ref2d[q * 2 + 1] * 50.f - 0.5f;
  float acc = 0.f;
  for (int k = 0; k < 2; ++k) {
    const float* vk = v + (size_t)k * NQ_ * EMBED_;
#pragma unroll
    for (int p = 0; p < 4; ++p) {
      int oi = q * 128 + ((k * 8 + h) * 4 + p) * 2;
      float x = rx + off[oi];
      float y = ry + off[oi + 1];
      float a = aw[q * 64 + (k * 8 + h) * 4 + p];
      acc += a * bilin_gather(vk, x, y, 50, 50, 0, h * 32 + lane);
    }
  }
  out[(size_t)q * EMBED_ + h * 32 + lane] = 0.5f * acc;
}

// one wave per (q, head); lane = head-dim
__global__ void sca_sample_kernel(const float* __restrict__ v,     // (6,5440,8,32)
                                  const float* __restrict__ off,   // (NQ,512)
                                  const float* __restrict__ aw,    // (NQ,256) softmaxed
                                  const float* __restrict__ refcam,// (6,NQ,4,2)
                                  const unsigned char* __restrict__ mask, // (6,NQ,4)
                                  float* __restrict__ slots)       // (NQ,256)
{
  int wid  = (blockIdx.x * blockDim.x + threadIdx.x) >> 5;
  int lane = threadIdx.x & 31;
  if (wid >= NQ_ * HEADS_) return;
  int q = wid >> 3, h = wid & 7;

  const int lw[4]     = {64, 32, 16, 8};
  const int lstart[4] = {0, 4096, 5120, 5376};

  float cnt = 0.f;
  float acc = 0.f;
  for (int cam = 0; cam < NCAM_; ++cam) {
    const unsigned char* mrow = mask + ((size_t)cam * NQ_ + q) * 4;
    bool hit = (mrow[0] | mrow[1] | mrow[2] | mrow[3]) != 0;
    if (!hit) continue;
    cnt += 1.f;
    const float* vc = v + (size_t)cam * TOT_ * EMBED_;
    const float* rc = refcam + ((size_t)cam * NQ_ + q) * 8;  // (4 z-anchors, 2)
#pragma unroll
    for (int l = 0; l < 4; ++l) {
      int Wl = lw[l];
#pragma unroll
      for (int p = 0; p < 8; ++p) {
        int dz = p & 3;
        int oi = q * 512 + ((h * 4 + l) * 8 + p) * 2;
        float x = rc[dz * 2 + 0] * (float)Wl + off[oi]     - 0.5f;
        float y = rc[dz * 2 + 1] * (float)Wl + off[oi + 1] - 0.5f;
        float a = aw[(size_t)q * 256 + h * 32 + l * 8 + p];
        acc += a * bilin_gather(vc, x, y, Wl, Wl, lstart[l], h * 32 + lane);
      }
    }
  }
  if (cnt < 1.f) cnt = 1.f;
  slots[(size_t)q * EMBED_ + h * 32 + lane] = acc / cnt;
}

// --------------------------------- driver -----------------------------------
static inline void launch_gemm(const float* A0, const float* A1, int split,
                               const float* W, const float* bias,
                               const float* resid, float* C,
                               int M, int N, int K, int relu, hipStream_t s)
{
  dim3 grid(N / BN, (M + BM - 1) / BM);
  gemm_wmma_f16<<<grid, 256, 0, s>>>(A0, A1, split, W, bias, resid, C, M, N, K, relu);
}

extern "C" void kernel_launch(void* const* d_in, const int* in_sizes, int n_in,
                              void* d_out, int out_size, void* d_ws, size_t ws_size,
                              hipStream_t stream)
{
  (void)in_sizes; (void)n_in; (void)out_size; (void)ws_size;

  const float* bevq   = (const float*)d_in[0];   // (NQ,256)
  const float* value  = (const float*)d_in[2];   // (6,5440,256)
  const float* pos    = (const float*)d_in[3];   // (NQ,256)
  const float* refcam = (const float*)d_in[6];   // (6,NQ,4,2)
  const unsigned char* mask = (const unsigned char*)d_in[7]; // (6,NQ,4) bool
  const float* ref2d  = (const float*)d_in[8];   // (NQ,2)
  const float* prev   = (const float*)d_in[9];   // (NQ,256)

  const float* tsa_off_w  = (const float*)d_in[10];
  const float* tsa_off_b  = (const float*)d_in[11];
  const float* tsa_attn_w = (const float*)d_in[12];
  const float* tsa_attn_b = (const float*)d_in[13];
  const float* tsa_val_w  = (const float*)d_in[14];
  const float* tsa_val_b  = (const float*)d_in[15];
  const float* tsa_out_w  = (const float*)d_in[16];
  const float* tsa_out_b  = (const float*)d_in[17];
  const float* sca_off_w  = (const float*)d_in[18];
  const float* sca_off_b  = (const float*)d_in[19];
  const float* sca_attn_w = (const float*)d_in[20];
  const float* sca_attn_b = (const float*)d_in[21];
  const float* sca_val_w  = (const float*)d_in[22];
  const float* sca_val_b  = (const float*)d_in[23];
  const float* sca_out_w  = (const float*)d_in[24];
  const float* sca_out_b  = (const float*)d_in[25];
  const float* ffn_w1     = (const float*)d_in[26];
  const float* ffn_b1     = (const float*)d_in[27];
  const float* ffn_w2     = (const float*)d_in[28];
  const float* ffn_b2     = (const float*)d_in[29];
  const float* ln_g       = (const float*)d_in[30];
  const float* ln_b       = (const float*)d_in[31];

  float* ws = (float*)d_ws;
  // workspace layout (floats); BIG1/BIG2 regions reused across phases
  float* scav    = ws;                       // 8,355,840  (6*5440*256)
  float* big1    = ws + 8355840;             // 1,280,000  qa | sca_off | ffn_h
  float* big2    = ws + 9635840;             // 1,280,000  tsa_v | (sca_attn, slots)
  float* tsaoff  = ws + 10915840;            //   320,000
  float* tsaaw   = ws + 11235840;            //   160,000
  float* tsas    = ws + 11395840;            //   640,000
  float* x1      = ws + 12035840;            //   640,000  post-TSA LN
  float* qp      = ws + 12675840;            //   640,000
  float* x3      = ws + 13315840;            //   640,000  post-SCA LN
  float* xt      = ws + 13955840;            //   640,000  pre-LN temp

  float* qa       = big1;
  float* tsa_v    = big2;
  float* sca_off  = big1;
  float* sca_attn = big2;
  float* slots    = big2 + 640000;
  float* ffn_h    = big1;
  float* out      = (float*)d_out;

  // ---------------- Temporal self-attention ----------------
  build_qa_kernel<<<(NQ_ * EMBED_ + 255) / 256, 256, 0, stream>>>(prev, bevq, pos, qa);
  // offsets & attention logits from qa (NQ x 512)
  launch_gemm(qa, qa, NQ_, tsa_off_w,  tsa_off_b,  nullptr, tsaoff, NQ_, 128, 512, 0, stream);
  launch_gemm(qa, qa, NQ_, tsa_attn_w, tsa_attn_b, nullptr, tsaaw,  NQ_,  64, 512, 0, stream);
  softmax_rows_kernel<<<(NQ_ * 16 + 255) / 256, 256, 0, stream>>>(tsaaw, NQ_ * 16, 4);
  // value projection of [prev; q]  (5000 x 256)
  launch_gemm(prev, bevq, NQ_, tsa_val_w, tsa_val_b, nullptr, tsa_v, 2 * NQ_, 256, 256, 0, stream);
  tsa_sample_kernel<<<(NQ_ * HEADS_) / 8, 256, 0, stream>>>(tsa_v, tsaoff, tsaaw, ref2d, tsas);
  // output projection + residual(q), then LN
  launch_gemm(tsas, tsas, NQ_, tsa_out_w, tsa_out_b, bevq, xt, NQ_, 256, 256, 0, stream);
  layernorm_kernel<<<(NQ_ + 7) / 8, 256, 0, stream>>>(xt, ln_g, ln_b, x1, NQ_);

  // ---------------- Spatial cross-attention ----------------
  add_kernel<<<(NQ_ * EMBED_ + 255) / 256, 256, 0, stream>>>(x1, pos, qp, NQ_ * EMBED_);
  launch_gemm(qp, qp, NQ_, sca_off_w,  sca_off_b,  nullptr, sca_off,  NQ_, 512, 256, 0, stream);
  launch_gemm(qp, qp, NQ_, sca_attn_w, sca_attn_b, nullptr, sca_attn, NQ_, 256, 256, 0, stream);
  softmax_rows_kernel<<<(NQ_ * HEADS_ + 255) / 256, 256, 0, stream>>>(sca_attn, NQ_ * HEADS_, 32);
  // per-camera value projection, one big GEMM (32640 x 256)
  launch_gemm(value, value, NCAM_ * TOT_, sca_val_w, sca_val_b, nullptr, scav,
              NCAM_ * TOT_, 256, 256, 0, stream);
  sca_sample_kernel<<<(NQ_ * HEADS_) / 8, 256, 0, stream>>>(scav, sca_off, sca_attn,
                                                            refcam, mask, slots);
  launch_gemm(slots, slots, NQ_, sca_out_w, sca_out_b, x1, xt, NQ_, 256, 256, 0, stream);
  layernorm_kernel<<<(NQ_ + 7) / 8, 256, 0, stream>>>(xt, ln_g, ln_b, x3, NQ_);

  // ---------------- FFN ----------------
  launch_gemm(x3, x3, NQ_, ffn_w1, ffn_b1, nullptr, ffn_h, NQ_, 512, 256, 1, stream);
  launch_gemm(ffn_h, ffn_h, NQ_, ffn_w2, ffn_b2, x3, xt, NQ_, 256, 512, 0, stream);
  layernorm_kernel<<<(NQ_ + 7) / 8, 256, 0, stream>>>(xt, ln_g, ln_b, out, NQ_);
}
